// RNN_42339787604736
// MI455X (gfx1250) — matrix operand order repaired and checked by reading, not compiled, not run on hardware
//
#include <hip/hip_runtime.h>
#include <hip/hip_bf16.h>

#define SEQ    512
#define BATCH  64
#define HID    1024
#define LAYERS 4

typedef __bf16 bf16_t;
typedef __attribute__((ext_vector_type(16))) __bf16 v16bf;
typedef __attribute__((ext_vector_type(8)))  __bf16 v8bf;
typedef __attribute__((ext_vector_type(4)))  __bf16 v4bf;
typedef __attribute__((ext_vector_type(8)))  float  v8f;

#define NT      16            // output columns per workgroup (one WMMA N-tile)
#define NWG     (HID / NT)    // 64 workgroups
#define THREADS 128           // 4 waves of 32
#define PITCHB  2064          // bytes per weight row in LDS: 2048 + 16 pad (16B aligned,
                              // 4-bank stride per row -> 16 lanes hit 16 distinct banks)

__device__ __forceinline__ v8f wmma_bf16(v16bf a, v16bf b, v8f c) {
  return __builtin_amdgcn_wmma_f32_16x16x32_bf16(false, a, false, b, (short)0, c,
                                                 false, false);
}

// One 16x16 tile, K=1024, A from global (bf16, row-major K-contiguous),
// B (= W rows, K-contiguous) from LDS. Accumulates into acc.
__device__ __forceinline__ v8f gemm16(const bf16_t* __restrict__ A,
                                      const char* ldsW,
                                      int mtile, int lane, v8f acc) {
  const int lm = lane & 15;   // A row within tile / B column
  const int kh = lane >> 4;   // K half-group select
  const bf16_t* arow = A + (size_t)(mtile * 16 + lm) * HID;
  const char*   wrow = ldsW + lm * PITCHB + kh * 16;
#pragma unroll 4
  for (int k0 = 0; k0 < HID; k0 += 32) {
    if ((k0 & 255) == 0)
      __builtin_prefetch(arow + k0 + 256, 0, 3);   // global_prefetch_b8
    // A 16x32 bf16 layout: lanes 0-15: K=k0..k0+7 (v0-3), K=k0+16..23 (v4-7);
    // lanes 16-31: +8 on both halves.
    v8bf alo = *(const v8bf*)(arow + k0 + kh * 8);
    v8bf ahi = *(const v8bf*)(arow + k0 + 16 + kh * 8);
    v16bf a = __builtin_shufflevector(alo, ahi, 0,1,2,3,4,5,6,7,8,9,10,11,12,13,14,15);
    // B 32x16 bf16 layout: lane (&15)=N, same K order as A (K=k0+8*kh..+7, k0+16+8*kh..+7).
    v8bf blo = *(const v8bf*)(wrow + k0 * 2);
    v8bf bhi = *(const v8bf*)(wrow + k0 * 2 + 32);
    v16bf b = __builtin_shufflevector(blo, bhi, 0,1,2,3,4,5,6,7,8,9,10,11,12,13,14,15);
    acc = wmma_bf16(a, b, acc);
  }
  return acc;
}

__global__ void __launch_bounds__(THREADS, 1)
rnn_persistent(const float* __restrict__ Wxh, const float* __restrict__ Whh,
               const float* __restrict__ bxh, const float* __restrict__ bhh,
               const bf16_t* __restrict__ xb, bf16_t* __restrict__ hb,
               float* __restrict__ out, unsigned* __restrict__ cnt) {
  extern __shared__ char lds[];
  const int tid  = threadIdx.x;
  const int lane = tid & 31;
  const int wave = tid >> 5;            // m-tile 0..3 (covers BATCH=64)
  const int n0   = blockIdx.x * NT;     // this WG's output columns

  // ---- one-time: convert this WG's weight columns fp32 -> bf16 into LDS ----
  for (int idx = tid; idx < LAYERS * 2 * NT * HID; idx += THREADS) {
    const int k   = idx & (HID - 1);
    const int n   = (idx >> 10) & (NT - 1);
    const int mat = (idx >> 14) & 1;
    const int l   = idx >> 15;
    const float* src = (mat ? Whh : Wxh) +
                       ((size_t)l * HID + (n0 + n)) * (size_t)HID + k;
    *(bf16_t*)(lds + ((l * 2 + mat) * NT + n) * PITCHB + k * 2) = (bf16_t)(*src);
  }
  float* lds_bias = (float*)(lds + LAYERS * 2 * NT * PITCHB);
  if (tid < LAYERS * NT) {
    const int l = tid >> 4, n = tid & 15;
    lds_bias[tid] = bxh[l * HID + n0 + n] + bhh[l * HID + n0 + n];
  }
  __syncthreads();

  const int ncol = n0 + (lane & 15);
  unsigned stage = 0;

  for (int t = 0; t < SEQ; ++t) {
    const int rp = t & 1;        // read phase of h state
    const int wp = rp ^ 1;       // write phase
    for (int l = 0; l < LAYERS; ++l) {
      const bf16_t* A1 = (l == 0)
          ? (xb + (size_t)t * BATCH * HID)
          : (hb + (size_t)(wp * LAYERS + (l - 1)) * BATCH * HID);
      const bf16_t* A2 = hb + (size_t)(rp * LAYERS + l) * BATCH * HID;
      const char* wx = lds + (l * 2 + 0) * NT * PITCHB;
      const char* wh = lds + (l * 2 + 1) * NT * PITCHB;

      const float bias = lds_bias[l * NT + (lane & 15)];
      v8f acc = {bias, bias, bias, bias, bias, bias, bias, bias};
      acc = gemm16(A1, wx, wave, lane, acc);   // x/W_xh^T (or prev layer h)
      acc = gemm16(A2, wh, wave, lane, acc);   // h/W_hh^T

      // tanh + writeback (C/D layout: N = lane&15, M = r + 8*(lane>=16))
      bf16_t* hdst = hb + (size_t)(wp * LAYERS + l) * BATCH * HID;
      float* odst = (l == LAYERS - 1) ? (out + (size_t)t * BATCH * HID) : nullptr;
      float* fdst = (t == SEQ - 1)
          ? (out + (size_t)SEQ * BATCH * HID + (size_t)l * BATCH * HID) : nullptr;
#pragma unroll
      for (int r = 0; r < 8; ++r) {
        const int m  = wave * 16 + r + ((lane >> 4) << 3);
        const float hv = tanhf(acc[r]);
        hdst[m * HID + ncol] = (bf16_t)hv;
        if (odst) odst[m * HID + ncol] = hv;
        if (fdst) fdst[m * HID + ncol] = hv;
      }

      // ---- device-wide barrier between dependent stages ----
      ++stage;
      __threadfence();           // make this thread's stores agent-visible
      __syncthreads();
      if (tid == 0) {
        __hip_atomic_fetch_add(cnt, 1u, __ATOMIC_ACQ_REL, __HIP_MEMORY_SCOPE_AGENT);
        const unsigned target = (unsigned)gridDim.x * stage;
        while (__hip_atomic_load(cnt, __ATOMIC_ACQUIRE, __HIP_MEMORY_SCOPE_AGENT) < target)
          __builtin_amdgcn_s_sleep(2);
      }
      __syncthreads();
    }
  }
}

// fp32 -> bf16 bulk conversion for x (one-time per launch; 33.5M elems)
__global__ void cvt_f32_to_bf16(const float* __restrict__ src,
                                bf16_t* __restrict__ dst, int n) {
  const int i = (blockIdx.x * blockDim.x + threadIdx.x) * 4;
  if (i >= n) return;
  const float4 v = *(const float4*)(src + i);
  v4bf o;
  o[0] = (bf16_t)v.x; o[1] = (bf16_t)v.y; o[2] = (bf16_t)v.z; o[3] = (bf16_t)v.w;
  *(v4bf*)(dst + i) = o;
}

// Initialize barrier counter and both phases of the bf16 h-state from h0.
__global__ void rnn_init(const float* __restrict__ h0, bf16_t* __restrict__ hb,
                         unsigned* __restrict__ cnt) {
  const int i = blockIdx.x * blockDim.x + threadIdx.x;
  if (i == 0) *cnt = 0u;
  const int n = LAYERS * BATCH * HID;
  if (i < n) {
    const bf16_t v = (bf16_t)h0[i];
    hb[i]     = v;    // phase 0 (read at t=0)
    hb[n + i] = v;    // phase 1 (overwritten before first read)
  }
}

extern "C" void kernel_launch(void* const* d_in, const int* in_sizes, int n_in,
                              void* d_out, int out_size, void* d_ws, size_t ws_size,
                              hipStream_t stream) {
  const float* x   = (const float*)d_in[0];
  const float* h0  = (const float*)d_in[1];
  const float* Wxh = (const float*)d_in[2];
  const float* Whh = (const float*)d_in[3];
  const float* bxh = (const float*)d_in[4];
  const float* bhh = (const float*)d_in[5];
  float* out = (float*)d_out;

  char* ws = (char*)d_ws;
  unsigned* cnt = (unsigned*)ws;                                   // 128 B slot
  bf16_t* hb = (bf16_t*)(ws + 128);                                // 2*L*B*H bf16 (1 MB)
  bf16_t* xb = (bf16_t*)(ws + 128 +
                         2ull * LAYERS * BATCH * HID * sizeof(bf16_t)); // 64 MB

  rnn_init<<<(LAYERS * BATCH * HID + 255) / 256, 256, 0, stream>>>(h0, hb, cnt);

  const int nx = SEQ * BATCH * HID;
  cvt_f32_to_bf16<<<(nx / 4 + 255) / 256, 256, 0, stream>>>(x, xb, nx);

  const size_t ldsBytes = (size_t)LAYERS * 2 * NT * PITCHB +
                          (size_t)LAYERS * NT * sizeof(float);     // ~264.4 KB of 320 KB
  rnn_persistent<<<NWG, THREADS, ldsBytes, stream>>>(Wxh, Whh, bxh, bhh,
                                                     xb, hb, out, cnt);
}
